// CrossAttentionBlock_68453188763714
// MI455X (gfx1250) — matrix operand order repaired and checked by reading, not compiled, not run on hardware
//
#include <hip/hip_runtime.h>
#include <hip/hip_bf16.h>

#define BB 4
#define CD 64
#define HH 256
#define WD 256
#define CP (CD + 8)     // padded row: 72 ushorts = 144B (16B aligned, bank-safe)
#define WP (WD + 8)     // padded row: 264 ushorts = 528B (16B aligned, bank-safe)

typedef __attribute__((ext_vector_type(16))) __bf16 bf16x16;
typedef __attribute__((ext_vector_type(8)))  float  f32x8;
typedef __attribute__((ext_vector_type(4)))  float  f32x4;
typedef __attribute__((ext_vector_type(4)))  unsigned int u32x4;

union Frag { bf16x16 h; unsigned int u[8]; };

// two f32 -> packed bf16x2 in one VALU op (RTNE in HW)
__device__ __forceinline__ unsigned int pk2(float a, float b) {
  unsigned int r;
  asm("v_cvt_pk_bf16_f32 %0, %1, %2" : "=v"(r) : "v"(a), "v"(b));
  return r;
}
// single f32 -> bf16 (low half of a pack)
__device__ __forceinline__ unsigned short f2bf(float f) {
  return (unsigned short)(pk2(f, 0.f) & 0xffffu);
}

// CDNA5 async memory->LDS DMA (no VGPR round-trip); tracked by ASYNCcnt.
__device__ __forceinline__ void async_ld_b128(unsigned lds_off, const void* gaddr) {
  asm volatile("global_load_async_to_lds_b128 %0, %1, off"
               :: "v"(lds_off), "v"(gaddr) : "memory");
}
__device__ __forceinline__ void wait_async0() {
  asm volatile("s_wait_asynccnt 0x0" ::: "memory");
}
__device__ __forceinline__ unsigned lds_off32(const void* p) {
  return (unsigned)(unsigned long long)p;   // generic LDS addr: offset in [31:0]
}

__device__ __forceinline__ f32x8 wmma_bf16(const Frag& a, const Frag& b, f32x8 c) {
  return __builtin_amdgcn_wmma_f32_16x16x32_bf16(false, a.h, false, b.h,
                                                 (short)0, c, false, false);
}

// ---------------------------------------------------------------------------
// Kernel 1: fused LayerNorm + q projection + (transposed) v projection, bf16.
// One workgroup per (stream, b, h).
//   q layout: [slice][w][c]   (rows are positions)   -> flash-attn Q/K
//   v layout: [slice][c][w]   (rows are channels)    -> flash-attn V (pre-transposed)
// ---------------------------------------------------------------------------
__global__ __launch_bounds__(256) void k1_ln_qv(
    const float* __restrict__ x,
    const float* __restrict__ wnA, const float* __restrict__ bnA,
    const float* __restrict__ wnB, const float* __restrict__ bnB,
    const float* __restrict__ W1A, const float* __restrict__ b1A,
    const float* __restrict__ W1B, const float* __restrict__ b1B,
    const float* __restrict__ W2A, const float* __restrict__ b2A,
    const float* __restrict__ W2B, const float* __restrict__ b2B,
    unsigned short* __restrict__ qout, unsigned short* __restrict__ vout)
{
  __shared__ unsigned short xb[WD][CP];   // normalized x, [w][c]
  __shared__ unsigned short vb[WD][CP];   // raw x,        [w][c]
  __shared__ unsigned short wq[CD][CP];   // W1 [o][c]
  __shared__ unsigned short wv[CD][CP];   // W2 [o][c]
  __shared__ float bq[CD], bv[CD], wns[CD], bns[CD];

  const int tid = threadIdx.x;
  const int blk = blockIdx.x;             // s*1024 + b*256 + h
  const int s   = blk >> 10;
  const int r0  = blk & 1023;
  const int b   = r0 >> 8;
  const int h   = r0 & 255;

  const float* W1 = s ? W1B : W1A;
  const float* W2 = s ? W2B : W2A;
  const float* b1 = s ? b1B : b1A;
  const float* b2 = s ? b2B : b2A;
  const float* wn = s ? wnB : wnA;
  const float* bn = s ? bnB : bnA;

  const float* xp = x + (((size_t)b * 128 + s * 64) * HH + h) * WD;

  // weights: convert in pairs -> one v_cvt_pk_bf16_f32 + dword LDS store
  for (int i = tid; i < (CD * CD) / 2; i += 256) {
    const int row = i >> 5;
    const int col = (i & 31) << 1;
    *(unsigned int*)&wq[row][col] = pk2(W1[2 * i], W1[2 * i + 1]);
    *(unsigned int*)&wv[row][col] = pk2(W2[2 * i], W2[2 * i + 1]);
  }
  if (tid < CD) { bq[tid] = b1[tid]; bv[tid] = b2[tid];
                  wns[tid] = wn[tid]; bns[tid] = bn[tid]; }

  // each thread owns position w = tid (coalesced channel-strided loads)
  float xv[CD];
  #pragma unroll
  for (int c = 0; c < CD; ++c) xv[c] = xp[(size_t)c * (HH * WD) + tid];

  float mu = 0.f;
  #pragma unroll
  for (int c = 0; c < CD; ++c) mu += xv[c];
  mu *= (1.f / 64.f);
  float var = 0.f;
  #pragma unroll
  for (int c = 0; c < CD; ++c) { float d = xv[c] - mu; var += d * d; }
  var *= (1.f / 64.f);
  const float rs = rsqrtf(var + 1e-6f);
  __syncthreads();   // wns/bns ready

  // write both bf16 rows contiguously (128-bit LDS stores)
  #pragma unroll
  for (int c0 = 0; c0 < CD; c0 += 8) {
    unsigned int px[4], pv[4];
    #pragma unroll
    for (int j = 0; j < 4; ++j) {
      const int c = c0 + 2 * j;
      px[j] = pk2((xv[c]   - mu) * rs * wns[c]   + bns[c],
                  (xv[c+1] - mu) * rs * wns[c+1] + bns[c+1]);
      pv[j] = pk2(xv[c], xv[c+1]);
    }
    *(u32x4*)&xb[tid][c0] = u32x4{px[0], px[1], px[2], px[3]};
    *(u32x4*)&vb[tid][c0] = u32x4{pv[0], pv[1], pv[2], pv[3]};
  }
  __syncthreads();

  const int wave = tid >> 5, lane = tid & 31;
  const int ln15 = lane & 15;
  const int half = lane >> 4;             // 0 / 1
  const int kbA  = half * 8;              // unused; A frags also use kbB
  const int kbB  = half * 16;             // B-frag K split

  // ------------------- q GEMM: D(m=o, n=w) --------------------------------
  {
    const int mt = wave & 3;
    const int n0 = (wave >> 2) * 8;
    Frag aq0, aq1;
    {
      const int m = mt * 16 + ln15;
      #pragma unroll
      for (int v2 = 0; v2 < 8; ++v2) {
        const int k = kbB + (v2 << 1);
        aq0.u[v2] = *(const unsigned int*)&wq[m][k];
        aq1.u[v2] = *(const unsigned int*)&wq[m][k + 32];
      }
    }
    unsigned short* qb = qout + (size_t)blk * (WD * CD);
    const int ob = mt * 16 + (half << 3);
    #pragma unroll
    for (int nt = n0; nt < n0 + 8; ++nt) {
      const int n = nt * 16 + ln15;       // position w
      Frag b0, b1f;
      #pragma unroll
      for (int v2 = 0; v2 < 8; ++v2) {    // contiguous -> ds_load_b128
        b0.u[v2]  = *(const unsigned int*)&xb[n][kbB + (v2 << 1)];
        b1f.u[v2] = *(const unsigned int*)&xb[n][kbB + (v2 << 1) + 32];
      }
      f32x8 acc = {0.f,0.f,0.f,0.f,0.f,0.f,0.f,0.f};
      acc = wmma_bf16(aq0, b0, acc);
      acc = wmma_bf16(aq1, b1f, acc);
      unsigned int pk[4];
      #pragma unroll
      for (int r = 0; r < 8; r += 2)
        pk[r >> 1] = pk2(acc[r] + bq[ob + r], acc[r + 1] + bq[ob + r + 1]);
      *(u32x4*)&qb[(size_t)n * CD + ob] = u32x4{pk[0], pk[1], pk[2], pk[3]};
    }
  }

  // ------------------- v GEMM (transposed): D(m=w, n=o) -------------------
  {
    unsigned short* vo = vout + (size_t)blk * (WD * CD);
    #pragma unroll
    for (int wi = 0; wi < 2; ++wi) {
      const int wt = wave * 2 + wi;
      Frag ax0, ax1;
      {
        const int m = wt * 16 + ln15;     // position w row of vb
        #pragma unroll
        for (int v2 = 0; v2 < 8; ++v2) {
          const int k = kbB + (v2 << 1);
          ax0.u[v2] = *(const unsigned int*)&vb[m][k];
          ax1.u[v2] = *(const unsigned int*)&vb[m][k + 32];
        }
      }
      const int wb2 = wt * 16 + (half << 3);
      #pragma unroll
      for (int ct = 0; ct < 4; ++ct) {
        const int n = ct * 16 + ln15;     // output channel o
        Frag b0, b1f;                     // B = W2^T : rows of wv
        #pragma unroll
        for (int v2 = 0; v2 < 8; ++v2) {
          b0.u[v2]  = *(const unsigned int*)&wv[n][kbB + (v2 << 1)];
          b1f.u[v2] = *(const unsigned int*)&wv[n][kbB + (v2 << 1) + 32];
        }
        f32x8 acc = {0.f,0.f,0.f,0.f,0.f,0.f,0.f,0.f};
        acc = wmma_bf16(ax0, b0, acc);
        acc = wmma_bf16(ax1, b1f, acc);
        const float bias = bv[n];
        unsigned int pk[4];
        #pragma unroll
        for (int r = 0; r < 8; r += 2)
          pk[r >> 1] = pk2(acc[r] + bias, acc[r + 1] + bias);
        *(u32x4*)&vo[(size_t)n * WD + wb2] = u32x4{pk[0], pk[1], pk[2], pk[3]};
      }
    }
  }
}

// ---------------------------------------------------------------------------
// Kernel 2: flash attention (seq 256, d 64) + beta/gamma scale + residual.
// One workgroup = 128 query rows of one (stream, b, h); 8 waves x 16 rows.
// fA = FA(qA, qB, vB); fB = FA(qB, qA, vA): stream s uses other stream's K/V.
// ---------------------------------------------------------------------------
__global__ __launch_bounds__(256) void k2_attn(
    const float* __restrict__ x,
    const unsigned short* __restrict__ q,
    const unsigned short* __restrict__ v,
    const float* __restrict__ beta,
    const float* __restrict__ gamma,
    float* __restrict__ out)
{
  __shared__ unsigned short Ks[WD][CP];       // keys   [x][c], padded rows
  __shared__ unsigned short Vt[CD][WP];       // values [c][x], padded rows
  __shared__ unsigned short Ps[8][16][40];    // per-wave P bounce, padded rows
  __shared__ float sc[CD];                    // beta or gamma

  const int tid  = threadIdx.x;
  const int blk  = blockIdx.x;                // bh*2 + rt
  const int rt   = blk & 1;
  const int bh   = blk >> 1;                  // s*1024 + b*256 + h
  const int s    = bh >> 10;
  const int r0   = bh & 1023;
  const int b    = r0 >> 8;
  const int h    = r0 & 255;
  const int obh  = s ? (bh - 1024) : (bh + 1024);

  const unsigned short* Qp = q + (size_t)bh  * (WD * CD);   // [w][c]
  const unsigned short* Kp = q + (size_t)obh * (WD * CD);   // [w][c]
  const unsigned short* Vp = v + (size_t)obh * (WD * CD);   // [c][w] (pre-transposed)

  // async DMA staging into padded LDS (no VGPR round-trip, ASYNCcnt-tracked)
  {
    const u32x4* ksrc = (const u32x4*)Kp;
    const u32x4* vsrc = (const u32x4*)Vp;
    #pragma unroll
    for (int i = tid; i < (WD * CD) / 8; i += 256) {
      async_ld_b128(lds_off32(&Ks[i >> 3][(i & 7) << 3]),  ksrc + i);
      async_ld_b128(lds_off32(&Vt[i >> 5][(i & 31) << 3]), vsrc + i);
    }
    if (tid < CD) sc[tid] = s ? gamma[tid] : beta[tid];
  }
  wait_async0();
  __syncthreads();

  const int wave = tid >> 5, lane = tid & 31;
  const int ln15 = lane & 15;
  const int half = lane >> 4;
  const int kbA  = half * 8;
  const int kbB  = half * 16;
  const int rowbase = rt * 128 + wave * 16;

  // Q A-fragments straight from global ([w][c] bf16)
  Frag qa0, qa1;
  {
    const int m = rowbase + ln15;
    #pragma unroll
    for (int v2 = 0; v2 < 8; ++v2) {
      const int k = kbB + (v2 << 1);
      qa0.u[v2] = *(const unsigned int*)&Qp[(size_t)m * CD + k];
      qa1.u[v2] = *(const unsigned int*)&Qp[(size_t)m * CD + k + 32];
    }
  }

  float m_i[8], l_i[8];
  f32x8 acc[4];
  #pragma unroll
  for (int r = 0; r < 8; ++r) { m_i[r] = -1e30f; l_i[r] = 0.f; }
  #pragma unroll
  for (int f = 0; f < 4; ++f) acc[f] = f32x8{0.f,0.f,0.f,0.f,0.f,0.f,0.f,0.f};

  const float scale = 0.125f;   // 64^-0.5

  for (int j = 0; j < 8; ++j) {           // 8 key tiles of 32
    const int jb = j * 32;

    // S = scale * Q @ K^T  (two 16x16 column sub-tiles)
    f32x8 S0 = {0.f,0.f,0.f,0.f,0.f,0.f,0.f,0.f};
    f32x8 S1 = {0.f,0.f,0.f,0.f,0.f,0.f,0.f,0.f};
    #pragma unroll
    for (int st = 0; st < 2; ++st) {
      const int xcol = jb + st * 16 + ln15;
      Frag kf0, kf1;
      #pragma unroll
      for (int v2 = 0; v2 < 8; ++v2) {    // contiguous -> ds_load_b128
        kf0.u[v2] = *(const unsigned int*)&Ks[xcol][kbB + (v2 << 1)];
        kf1.u[v2] = *(const unsigned int*)&Ks[xcol][kbB + (v2 << 1) + 32];
      }
      if (st == 0) { S0 = wmma_bf16(qa0, kf0, S0); S0 = wmma_bf16(qa1, kf1, S0); }
      else         { S1 = wmma_bf16(qa0, kf0, S1); S1 = wmma_bf16(qa1, kf1, S1); }
    }
    #pragma unroll
    for (int r = 0; r < 8; ++r) { S0[r] *= scale; S1[r] *= scale; }

    // online softmax: row stats across the 16-lane half owning each row
    float alpha[8];
    #pragma unroll
    for (int r = 0; r < 8; ++r) {
      float rm = fmaxf(S0[r], S1[r]);
      rm = fmaxf(rm, __shfl_xor(rm, 1));
      rm = fmaxf(rm, __shfl_xor(rm, 2));
      rm = fmaxf(rm, __shfl_xor(rm, 4));
      rm = fmaxf(rm, __shfl_xor(rm, 8));
      const float mn = fmaxf(m_i[r], rm);
      alpha[r] = __expf(m_i[r] - mn);
      m_i[r] = mn;
      S0[r] = __expf(S0[r] - mn);
      S1[r] = __expf(S1[r] - mn);
      float rsum = S0[r] + S1[r];
      rsum += __shfl_xor(rsum, 1);
      rsum += __shfl_xor(rsum, 2);
      rsum += __shfl_xor(rsum, 4);
      rsum += __shfl_xor(rsum, 8);
      l_i[r] = l_i[r] * alpha[r] + rsum;
    }
    #pragma unroll
    for (int f = 0; f < 4; ++f)
      #pragma unroll
      for (int r = 0; r < 8; ++r) acc[f][r] *= alpha[r];

    // bounce P (D-layout) -> A-fragment layout; Ps is wave-private, so
    // in-wave s_wait_dscnt ordering (compiler-inserted) is sufficient.
    {
      const int mrow = half << 3;
      #pragma unroll
      for (int r = 0; r < 8; ++r) {
        Ps[wave][mrow + r][ln15]      = f2bf(S0[r]);
        Ps[wave][mrow + r][ln15 + 16] = f2bf(S1[r]);
      }
    }
    Frag pa;
    #pragma unroll
    for (int v2 = 0; v2 < 8; ++v2) {
      const int k = kbB + (v2 << 1);
      pa.u[v2] = *(const unsigned int*)&Ps[wave][ln15][k];
    }

    // O += P @ V  (4 channel tiles of 16); V rows contiguous -> ds_load_b128
    #pragma unroll
    for (int f = 0; f < 4; ++f) {
      const int n = f * 16 + ln15;        // channel
      Frag vf;
      #pragma unroll
      for (int v2 = 0; v2 < 8; ++v2)
        vf.u[v2] = *(const unsigned int*)&Vt[n][jb + kbB + (v2 << 1)];
      acc[f] = wmma_bf16(pa, vf, acc[f]);
    }
  }

  // finalize: out = x + sc[c] * O / l   (contiguous along w -> f32x4 stores)
  float rl[8];
  #pragma unroll
  for (int r = 0; r < 8; ++r) rl[r] = 1.f / l_i[r];
  const int wb2 = rowbase + (half << 3);
  #pragma unroll
  for (int f = 0; f < 4; ++f) {
    const int c = f * 16 + ln15;
    const size_t gidx = (((size_t)b * 128 + s * 64 + c) * HH + h) * WD + wb2;
    const f32x4 x0 = *(const f32x4*)(x + gidx);
    const f32x4 x1 = *(const f32x4*)(x + gidx + 4);
    const float scv = sc[c];
    f32x4 o0, o1;
    #pragma unroll
    for (int r = 0; r < 4; ++r) {
      o0[r] = x0[r] + scv * acc[f][r]     * rl[r];
      o1[r] = x1[r] + scv * acc[f][r + 4] * rl[r + 4];
    }
    *(f32x4*)(out + gidx)     = o0;
    *(f32x4*)(out + gidx + 4) = o1;
  }
}

extern "C" void kernel_launch(void* const* d_in, const int* in_sizes, int n_in,
                              void* d_out, int out_size, void* d_ws, size_t ws_size,
                              hipStream_t stream) {
  (void)in_sizes; (void)n_in; (void)out_size; (void)ws_size;
  const float* x    = (const float*)d_in[0];
  const float* wnA  = (const float*)d_in[1];
  const float* bnA  = (const float*)d_in[2];
  const float* wnB  = (const float*)d_in[3];
  const float* bnB  = (const float*)d_in[4];
  const float* W1A  = (const float*)d_in[5];
  const float* b1A  = (const float*)d_in[6];
  const float* W1B  = (const float*)d_in[7];
  const float* b1B  = (const float*)d_in[8];
  const float* W2A  = (const float*)d_in[9];
  const float* b2A  = (const float*)d_in[10];
  const float* W2B  = (const float*)d_in[11];
  const float* b2B  = (const float*)d_in[12];
  const float* beta = (const float*)d_in[13];
  const float* gamma= (const float*)d_in[14];

  unsigned short* qbuf = (unsigned short*)d_ws;                       // 64 MB
  unsigned short* vbuf = qbuf + (size_t)2 * BB * HH * WD * CD;        // 64 MB

  k1_ln_qv<<<2 * BB * HH, 256, 0, stream>>>(x, wnA, bnA, wnB, bnB,
                                            W1A, b1A, W1B, b1B,
                                            W2A, b2A, W2B, b2B, qbuf, vbuf);
  k2_attn<<<2 * BB * HH * 2, 256, 0, stream>>>(x, qbuf, vbuf, beta, gamma,
                                               (float*)d_out);
}
